// FastSelfAttn_80539226735289
// MI455X (gfx1250) — matrix operand order repaired and checked by reading, not compiled, not run on hardware
//
#include <hip/hip_runtime.h>

// Problem constants (B=2, S=2048, D=2048, H=16, DH=128)
#define B_SZ    2
#define S_LEN   2048
#define D_MODEL 2048
#define NHEAD   16
#define DHEAD   128
#define N3D     6144   // 3*D

typedef __attribute__((ext_vector_type(16))) __bf16       v16bf;
typedef __attribute__((ext_vector_type(8)))  float        v8f;
typedef __attribute__((ext_vector_type(4)))  unsigned int u32x4;
typedef unsigned short u16;

union FragB16 { u32x4 q[2]; v16bf v; };   // 32 bytes = 16 bf16 = one WMMA A/B fragment per lane

// round-to-nearest-even f32 -> bf16
__device__ __forceinline__ u16 f2bf(float f) {
  unsigned int u = __float_as_uint(f);
  u += 0x7fffu + ((u >> 16) & 1u);
  return (u16)(u >> 16);
}

__device__ __forceinline__ v8f wmma_bf(v16bf a, v16bf b, v8f c) {
  // (neg_a, A, neg_b, B, c_mod, C, reuse_a, reuse_b)
  return __builtin_amdgcn_wmma_f32_16x16x32_bf16(false, a, false, b, (short)0, c, false, false);
}

// Async global->LDS copy of 16 bytes per lane (GLOBAL_LOAD_ASYNC_TO_LDS_B128,
// GV mode: 64-bit per-lane global address, 32-bit per-lane LDS byte offset).
// For __shared__ pointers the LDS byte offset is the low 32 bits of the flat
// address (the shared aperture occupies addr[63:32] only).
__device__ __forceinline__ void async_copy_b128(void* lds_dst, const void* gsrc) {
  unsigned dst = (unsigned)(size_t)lds_dst;
  unsigned long long src = (unsigned long long)(size_t)gsrc;
  asm volatile("global_load_async_to_lds_b128 %0, %1, off"
               :: "v"(dst), "v"(src) : "memory");
}
__device__ __forceinline__ void async_wait0() {
  asm volatile("s_wait_asynccnt 0x0" ::: "memory");
}

// ---------------------------------------------------------------------------
// Kernel 1: qkv = X @ W + b   (M=4096, K=2048, N=6144), output bf16 (ushort)
// block tile 128x128, 8 waves arranged 4(M) x 2(N), each wave 32x64, K-step 32
// ---------------------------------------------------------------------------
__global__ __launch_bounds__(256) void qkv_gemm_kernel(const float* __restrict__ X,
                                                       const float* __restrict__ W,
                                                       const float* __restrict__ bias,
                                                       u16* __restrict__ qkv) {
  __shared__ __align__(16) u16 aS[128][40];   // [m][k]  (pitch-padded)
  __shared__ __align__(16) u16 bS[128][40];   // [n][k]  (W tile transposed)

  const int tid  = threadIdx.x;
  const int wave = tid >> 5;
  const int lane = tid & 31;
  const int lm   = lane & 15;
  const int half = lane >> 4;
  const int m0   = blockIdx.y * 128;
  const int n0   = blockIdx.x * 128;
  const int mw   = (wave >> 1) * 32;   // wave M offset in block tile
  const int nw   = (wave & 1) * 64;    // wave N offset in block tile

  v8f acc[2][4] = {};

  for (int k0 = 0; k0 < D_MODEL; k0 += 32) {
    __syncthreads();
    // A tile: 128x32 f32 -> bf16 LDS (row-major)
#pragma unroll
    for (int i = 0; i < 4; ++i) {
      int e = tid + i * 256;                 // 1024 float4s
      int m = e >> 3;                        // 8 float4 per 32-wide row
      int c = (e & 7) * 4;
      const float4 x = *(const float4*)(X + (size_t)(m0 + m) * D_MODEL + k0 + c);
      unsigned long long pk = (unsigned long long)f2bf(x.x)
                            | ((unsigned long long)f2bf(x.y) << 16)
                            | ((unsigned long long)f2bf(x.z) << 32)
                            | ((unsigned long long)f2bf(x.w) << 48);
      *(unsigned long long*)&aS[m][c] = pk;
    }
    // W tile: 32x128 f32, stored transposed -> bS[n][k]
#pragma unroll
    for (int i = 0; i < 4; ++i) {
      int e  = tid + i * 256;
      int kk = e >> 5;                       // 32 float4 per 128-wide row
      int c  = (e & 31) * 4;
      const float4 x = *(const float4*)(W + (size_t)(k0 + kk) * N3D + n0 + c);
      bS[c + 0][kk] = f2bf(x.x);
      bS[c + 1][kk] = f2bf(x.y);
      bS[c + 2][kk] = f2bf(x.z);
      bS[c + 3][kk] = f2bf(x.w);
    }
    // prefetch next K-step tiles into cache while this step computes
    if (k0 + 32 < D_MODEL) {
      __builtin_prefetch(X + (size_t)(m0 + (tid >> 1)) * D_MODEL + k0 + 32 + (tid & 1) * 16, 0, 0);
      __builtin_prefetch(W + (size_t)(k0 + 32 + (tid >> 3)) * N3D + n0 + (tid & 7) * 16, 0, 0);
    }
    __syncthreads();

    // A fragments: lane<16 -> M=lm, K 0..7 & 16..23 ; lane>=16 -> K 8..15 & 24..31
    FragB16 af[2];
#pragma unroll
    for (int mi = 0; mi < 2; ++mi) {
      const u16* ab = &aS[mw + mi * 16 + lm][half * 8];
      af[mi].q[0] = *(const u32x4*)ab;
      af[mi].q[1] = *(const u32x4*)(ab + 16);
    }
    // B fragments: lane<16 -> N=lm, K 0..7 & 16..23 ; lane>=16 -> K 8..15 & 24..31
#pragma unroll
    for (int ni = 0; ni < 4; ++ni) {
      FragB16 bf;
      const u16* bb = &bS[nw + ni * 16 + lm][half * 8];
      bf.q[0] = *(const u32x4*)bb;
      bf.q[1] = *(const u32x4*)(bb + 16);
#pragma unroll
      for (int mi = 0; mi < 2; ++mi)
        acc[mi][ni] = wmma_bf(af[mi].v, bf.v, acc[mi][ni]);
    }
  }

  // epilogue: + bias, convert to bf16, store
#pragma unroll
  for (int mi = 0; mi < 2; ++mi)
#pragma unroll
    for (int ni = 0; ni < 4; ++ni) {
      int N = n0 + nw + ni * 16 + lm;
      float bv = bias[N];
#pragma unroll
      for (int r = 0; r < 8; ++r) {
        int M = m0 + mw + mi * 16 + r + 8 * half;  // C layout: M = r + 8*(lane/16)
        qkv[(size_t)M * N3D + N] = f2bf(acc[mi][ni][r] + bv);
      }
    }
}

// ---------------------------------------------------------------------------
// The faithful flat reshape [B,S,D]->[B,H,S,DH] means attention row (b,h,s)
// lives at qkv row (b*2048 + h*128 + s/16), col (mat*2048 + (s%16)*128),
// as 128 contiguous bf16.
// ---------------------------------------------------------------------------
__device__ __forceinline__ const u16* qkv_row(const u16* qkv, int mat, int b, int h, int s) {
  int row = b * S_LEN + h * 128 + (s >> 4);
  int col = mat * D_MODEL + (s & 15) * DHEAD;
  return qkv + (size_t)row * N3D + col;
}

// ---------------------------------------------------------------------------
// Kernel 2: flash attention per (b,h,128-row q tile). 8 waves x 16 q rows.
// Mask is j >= i -> -1e9 (diagonal masked). All KV tiles visited so the
// fully-masked row 0 reproduces the reference's uniform softmax exactly.
// K tile is staged via async global->LDS (ASYNCcnt); V tile is transposed
// through VGPRs so P@V B-fragments are contiguous ds_load_b128.
// ---------------------------------------------------------------------------
__global__ __launch_bounds__(256) void flash_attn_kernel(const u16* __restrict__ qkv,
                                                         float* __restrict__ out) {
  __shared__ __align__(16) u16 kS[128][136];      // K tile, [kv][d]
  __shared__ __align__(16) u16 vT[128][136];      // V tile transposed, [d][kv]
  __shared__ __align__(16) u16 pS[8][16][136];    // per-wave P staging (C->A relayout)

  const int tid  = threadIdx.x;
  const int wave = tid >> 5;
  const int lane = tid & 31;
  const int lm   = lane & 15;
  const int half = lane >> 4;
  const int qt   = blockIdx.x;
  const int bh   = blockIdx.y;
  const int b    = bh >> 4;
  const int h    = bh & 15;
  const int q_base = qt * 128 + wave * 16;

  // Preload this wave's q fragments (constant across all KV tiles): 4 K-chunks of 32
  FragB16 qf[4];
  {
    const u16* qr = qkv_row(qkv, 0, b, h, q_base + lm);
#pragma unroll
    for (int kk = 0; kk < 4; ++kk) {
      qf[kk].q[0] = *(const u32x4*)(qr + kk * 32 + half * 8);
      qf[kk].q[1] = *(const u32x4*)(qr + kk * 32 + 16 + half * 8);
    }
  }

  v8f o[8] = {};
  float m_run[8], l_run[8];
#pragma unroll
  for (int r = 0; r < 8; ++r) { m_run[r] = -__builtin_inff(); l_run[r] = 0.0f; }

  const float SCALE = 0.08838834764831845f;  // 1/sqrt(128)

  for (int kt = 0; kt < S_LEN / 128; ++kt) {
    __syncthreads();
    // K tile: async global->LDS, one b128 per lane per issue (no VGPR round trip)
#pragma unroll
    for (int i = 0; i < 8; ++i) {
      int e  = tid + i * 256;                // 2048 16-byte chunks
      int kv = e >> 4;
      int c  = (e & 15) * 8;
      const u16* kr = qkv_row(qkv, 1, b, h, kt * 128 + kv) + c;
      async_copy_b128(&kS[kv][c], kr);
    }
    // V tile: load rows, scatter transposed into vT[d][kv]
    for (int e = tid; e < 2048; e += 256) {
      int kv = e >> 4;
      int c  = (e & 15) * 8;
      const u16* vr = qkv_row(qkv, 2, b, h, kt * 128 + kv);
      u32x4 x = *(const u32x4*)(vr + c);
      const u16* xs = (const u16*)&x;
#pragma unroll
      for (int j = 0; j < 8; ++j) vT[c + j][kv] = xs[j];
    }
    async_wait0();
    __syncthreads();

    // --- scores: 16x128 = q(16x128) @ k^T ---
    v8f sc[8];
#pragma unroll
    for (int nt = 0; nt < 8; ++nt) {
      v8f c = {};
#pragma unroll
      for (int kk = 0; kk < 4; ++kk) {
        FragB16 bf;   // B frag: lane -> N = kv col (row of kS), K = d in table order
        const u16* kb = &kS[nt * 16 + lm][kk * 32 + half * 8];
        bf.q[0] = *(const u32x4*)kb;
        bf.q[1] = *(const u32x4*)(kb + 16);
        c = wmma_bf(qf[kk].v, bf.v, c);
      }
      sc[nt] = c;
    }

    // --- scale + causal mask (j >= i => -1e9) in C layout ---
#pragma unroll
    for (int nt = 0; nt < 8; ++nt) {
      int j = kt * 128 + nt * 16 + lm;
#pragma unroll
      for (int r = 0; r < 8; ++r) {
        int i = q_base + r + 8 * half;
        float v = sc[nt][r] * SCALE;
        sc[nt][r] = (j >= i) ? -1.0e9f : v;
      }
    }

    // --- online softmax: row max across 8 tiles + 16-lane half-wave reduce ---
    float rmax[8];
#pragma unroll
    for (int r = 0; r < 8; ++r) {
      float mx = sc[0][r];
#pragma unroll
      for (int nt = 1; nt < 8; ++nt) mx = fmaxf(mx, sc[nt][r]);
#pragma unroll
      for (int off = 1; off < 16; off <<= 1) mx = fmaxf(mx, __shfl_xor(mx, off, 32));
      rmax[r] = mx;
    }

    float corr[8];
#pragma unroll
    for (int r = 0; r < 8; ++r) {
      float mn = fmaxf(m_run[r], rmax[r]);
      corr[r]  = __expf(m_run[r] - mn);
      m_run[r] = mn;
    }

    // p = exp(s - m); accumulate row sums; stage p (bf16) into per-wave LDS
    float rs[8];
#pragma unroll
    for (int r = 0; r < 8; ++r) rs[r] = 0.0f;
#pragma unroll
    for (int nt = 0; nt < 8; ++nt) {
#pragma unroll
      for (int r = 0; r < 8; ++r) {
        float p = __expf(sc[nt][r] - m_run[r]);
        rs[r] += p;
        pS[wave][r + 8 * half][nt * 16 + lm] = f2bf(p);
      }
    }
#pragma unroll
    for (int r = 0; r < 8; ++r) {
      float s = rs[r];
#pragma unroll
      for (int off = 1; off < 16; off <<= 1) s += __shfl_xor(s, off, 32);
      l_run[r] = l_run[r] * corr[r] + s;
    }

    // rescale running output
#pragma unroll
    for (int dt = 0; dt < 8; ++dt)
#pragma unroll
      for (int r = 0; r < 8; ++r) o[dt][r] *= corr[r];

    // --- o += p @ v : A = p (from per-wave LDS), B = v (from transposed LDS) ---
    FragB16 pf[4];
#pragma unroll
    for (int kc = 0; kc < 4; ++kc) {
      const u16* pb = &pS[wave][lm][kc * 32 + half * 8];
      pf[kc].q[0] = *(const u32x4*)pb;
      pf[kc].q[1] = *(const u32x4*)(pb + 16);
    }
#pragma unroll
    for (int dt = 0; dt < 8; ++dt) {
#pragma unroll
      for (int kc = 0; kc < 4; ++kc) {
        FragB16 vf;   // B frag: lane -> N = d (row of vT), K = kv in table order
        const u16* vb = &vT[dt * 16 + lm][kc * 32 + half * 8];
        vf.q[0] = *(const u32x4*)vb;
        vf.q[1] = *(const u32x4*)(vb + 16);
        o[dt] = wmma_bf(pf[kc].v, vf.v, o[dt]);
      }
    }
  }

  // epilogue: normalize by l and store f32 to flat [B,S,D] (= [B,H,S,DH] flat)
#pragma unroll
  for (int r = 0; r < 8; ++r) {
    float inv = 1.0f / l_run[r];
    int srow  = q_base + r + 8 * half;
    size_t ob = ((size_t)(b * NHEAD + h) * S_LEN + srow) * DHEAD;
#pragma unroll
    for (int dt = 0; dt < 8; ++dt)
      out[ob + dt * 16 + lm] = o[dt][r] * inv;
  }
}

// ---------------------------------------------------------------------------
extern "C" void kernel_launch(void* const* d_in, const int* in_sizes, int n_in,
                              void* d_out, int out_size, void* d_ws, size_t ws_size,
                              hipStream_t stream) {
  (void)in_sizes; (void)n_in; (void)out_size; (void)ws_size;
  const float* X    = (const float*)d_in[0];   // query_vector [2,2048,2048]
  // d_in[1], d_in[2] (key/value vectors) are unused by the reference forward.
  const float* W    = (const float*)d_in[3];   // [2048, 6144]
  const float* bias = (const float*)d_in[4];   // [6144]
  u16*  qkv = (u16*)d_ws;                      // bf16 qkv scratch: 4096*6144*2 = 50.3 MB
  float* out = (float*)d_out;

  dim3 g1(N3D / 128, (B_SZ * S_LEN) / 128);    // 48 x 32
  qkv_gemm_kernel<<<g1, 256, 0, stream>>>(X, W, bias, qkv);

  dim3 g2(S_LEN / 128, B_SZ * NHEAD);          // 16 x 32
  flash_attn_kernel<<<g2, 256, 0, stream>>>(qkv, out);
}
